// KSCO_1_72756745994336
// MI455X (gfx1250) — matrix operand order repaired and checked by reading, not compiled, not run on hardware
//
#include <hip/hip_runtime.h>
#include <hip/hip_bf16.h>
#include <math.h>

// ---------------------------------------------------------------------------
// MI455X (gfx1250) implementation. wave32, WMMA bf16 16x16x32 with f32 accum.
// A operands staged in LDS in the ISA-documented A layout (lane&15 = M row,
// two 8-elem K groups at +0/+16); B operands read as B^T rows (lane&15 = N,
// same two 8-elem K groups at +0/+16 as A) directly from L2-resident bf16 weights.
// conv1 weights are permuted to K' = tap*256 + cin so every 32-wide K chunk
// has a fixed filter tap -> division-free, loop-invariant im2col addressing.
// ---------------------------------------------------------------------------

typedef __attribute__((ext_vector_type(16))) __bf16 v16bf;
typedef __attribute__((ext_vector_type(8)))  __bf16 v8bf;
typedef __attribute__((ext_vector_type(8)))  float  v8f;

static __device__ __forceinline__ v8f wmma_bf16(v16bf a, v16bf b, v8f c) {
  return __builtin_amdgcn_wmma_f32_16x16x32_bf16(false, a, false, b, (short)0, c,
                                                 false, false);
}

static __device__ __forceinline__ v16bf cat16(v8bf lo, v8bf hi) {
  v16bf r;
#pragma unroll
  for (int i = 0; i < 8; ++i) { r[i] = lo[i]; r[i + 8] = hi[i]; }
  return r;
}

// A operand: 16x32 (MxK) tile from row-major [m][k] buffer (stride ld elems).
static __device__ __forceinline__ v16bf load_a(const __bf16* base, int ld,
                                               int m0, int k0, int lane) {
  int m  = m0 + (lane & 15);
  int kb = k0 + ((lane >> 4) << 3);           // 0 or 8
  const __bf16* p = base + m * ld + kb;
  v8bf lo = *(const v8bf*)p;                  // K = kb .. kb+7
  v8bf hi = *(const v8bf*)(p + 16);           // K = kb+16 .. kb+23
  return cat16(lo, hi);
}

// B operand: 32x16 (KxN) tile, sourced from N-major [n][k] buffer (B^T rows).
static __device__ __forceinline__ v16bf load_bt(const __bf16* base, int ld,
                                                int n0, int k0, int lane) {
  int n  = n0 + (lane & 15);
  int kb = k0 + ((lane >> 4) << 3);           // 0 or 8
  const __bf16* p = base + n * ld + kb;
  v8bf lo = *(const v8bf*)p;                  // K = kb .. kb+7
  v8bf hi = *(const v8bf*)(p + 16);           // K = kb+16 .. kb+23
  return cat16(lo, hi);
}

// ---------------------------------------------------------------------------
// prep: f32 -> bf16 weight conversion (conv1 K-permuted), zero accumulators
// ---------------------------------------------------------------------------
__global__ __launch_bounds__(256) void k_prep(
    const float* __restrict__ c1w, const float* __restrict__ qw,
    const float* __restrict__ pw,
    __bf16* __restrict__ c1b, __bf16* __restrict__ qb, __bf16* __restrict__ pb,
    float* __restrict__ s_sum, float* __restrict__ sta_sum) {
  int i = blockIdx.x * 256 + threadIdx.x;
  if (i < 294912) {
    // permuted layout: c1b[o][r*256 + c] = conv1_w[o][c*9 + r]
    int o  = i / 2304;
    int kp = i - o * 2304;
    int r  = kp >> 8;          // filter tap 0..8
    int c  = kp & 255;         // input channel
    c1b[i] = (__bf16)c1w[o * 2304 + c * 9 + r];
  }
  if (i < 49152) qb[i] = (__bf16)qw[i];
  if (i < 16384) pb[i] = (__bf16)pw[i];
  if (i < 4096) { s_sum[i] = 0.f; sta_sum[i] = 0.f; }
}

// ---------------------------------------------------------------------------
// conv1 3x3 (256->128) implicit GEMM + BN + LeakyReLU, fused LN tokens + cs.
// Block = 128 threads (4 waves). M-tile = 32 positions (one row of w),
// N = 128 out channels (32 per wave), K' = 2304 in 72 chunks of 32 (tap-major,
// so the gather address is base + c*9216 with no divisions). A tile is
// double-buffered with a software pipeline: one barrier per chunk, global
// loads for chunk i+1 in flight while WMMAs consume chunk i.
// ---------------------------------------------------------------------------
__global__ __launch_bounds__(128) void k_conv1(
    const float* __restrict__ x, const __bf16* __restrict__ wb,
    const float* __restrict__ bg, const float* __restrict__ bb,
    const float* __restrict__ bm, const float* __restrict__ bv,
    const float* __restrict__ lng, const float* __restrict__ lnb,
    const float* __restrict__ c2w,
    __bf16* __restrict__ lnt, float* __restrict__ cs) {
  __shared__ __align__(32) __bf16 At[2][32 * 32];  // double-buffered A tile
  __shared__ __align__(32) float  xt[32 * 132];    // activated outputs, padded
  __shared__ float smu[32], srs[32];

  int blk = blockIdx.x;                // 4608 blocks
  int b   = blk / 288;                 // 288 M-tiles per image
  int rem = blk - b * 288;
  int h   = rem / 3;
  int w0  = (rem - h * 3) * 32;
  int tid = threadIdx.x;
  int lane = tid & 31, wid = tid >> 5;
  int nbase = wid * 32;
  int ln15 = lane & 15;
  int rbase = (lane >> 4) << 3;

  v8f z8 = {0.f, 0.f, 0.f, 0.f, 0.f, 0.f, 0.f, 0.f};
  v8f acc[2][2] = {{z8, z8}, {z8, z8}};

  int mldr = tid & 31;                 // position (w offset) this thread stages
  int ksub = tid >> 5;                 // which 8-wide K group (0..3)
  int xb   = b * 2359296;              // b * 256 * 96 * 96

  float regs[8];
  // stage chunk 'ch' (r = ch>>3, c0 = (ch&7)*32) into regs
  auto stage_load = [&](int ch) {
    int r  = ch >> 3;
    int c0 = (ch & 7) << 5;
    int kh = r / 3;                      // loop-invariant per chunk (scalar)
    int kw = r - kh * 3;
    int hh = h + kh - 1;
    int wwp = w0 + mldr + kw - 1;
    bool ok = ((unsigned)hh < 96u) & ((unsigned)wwp < 96u);
    int base = xb + (c0 + ksub * 8) * 9216 + hh * 96 + wwp;
#pragma unroll
    for (int j = 0; j < 8; ++j)
      regs[j] = ok ? x[base + j * 9216] : 0.f;
  };
  auto stage_store = [&](int p) {
    v8bf vv;
#pragma unroll
    for (int j = 0; j < 8; ++j) vv[j] = (__bf16)regs[j];
    *(v8bf*)(&At[p][mldr * 32 + ksub * 8]) = vv;
  };

  stage_load(0);
  for (int ch = 0; ch < 72; ++ch) {
    int p = ch & 1;
    stage_store(p);
    if (ch + 1 < 72) stage_load(ch + 1);   // loads in flight during WMMAs
    __syncthreads();
    int k0 = ((ch >> 3) << 8) + ((ch & 7) << 5);  // permuted-K chunk base
    v16bf a0 = load_a(At[p], 32, 0, 0, lane);
    v16bf a1 = load_a(At[p], 32, 16, 0, lane);
#pragma unroll
    for (int nt = 0; nt < 2; ++nt) {
      v16bf bt = load_bt(wb, 2304, nbase + nt * 16, k0, lane);
      acc[0][nt] = wmma_bf16(a0, bt, acc[0][nt]);
      acc[1][nt] = wmma_bf16(a1, bt, acc[1][nt]);
    }
  }
  __syncthreads();

  // epilogue: BN + LeakyReLU into LDS
#pragma unroll
  for (int mt = 0; mt < 2; ++mt)
#pragma unroll
    for (int nt = 0; nt < 2; ++nt) {
      int ch = nbase + nt * 16 + ln15;
      float scale = bg[ch] * rsqrtf(bv[ch] + 1e-5f);
      float shift = bb[ch] - bm[ch] * scale;
#pragma unroll
      for (int i = 0; i < 8; ++i) {
        int mm = mt * 16 + rbase + i;
        float y = acc[mt][nt][i] * scale + shift;
        y = y > 0.f ? y : 0.01f * y;
        xt[mm * 132 + ch] = y;
      }
    }
  __syncthreads();
  // per-token LN stats + cs = sigmoid(conv2 . x1)
  if (tid < 32) {
    float s = 0.f, ss = 0.f, dot = 0.f;
    for (int c = 0; c < 128; ++c) {
      float yv = xt[tid * 132 + c];
      s += yv; ss += yv * yv; dot += c2w[c] * yv;
    }
    float mu  = s * (1.f / 128.f);
    float var = ss * (1.f / 128.f) - mu * mu;
    smu[tid] = mu;
    srs[tid] = rsqrtf(var + 1e-6f);
    int n = h * 96 + w0 + tid;
    cs[b * 9216 + n] = 1.f / (1.f + __expf(-dot));
  }
  __syncthreads();
  // write LN'd tokens as bf16 (B, 9216, 128)
  for (int e = tid; e < 4096; e += 128) {
    int mm = e >> 7;
    int c  = e & 127;
    float y = (xt[mm * 132 + c] - smu[mm]) * srs[mm] * lng[c] + lnb[c];
    int n = h * 96 + w0 + mm;
    lnt[((size_t)b * 9216 + n) * 128 + c] = (__bf16)y;
  }
}

// ---------------------------------------------------------------------------
// window attention: one 7x7 window (49 tokens, pad to 64) per 256-thread block
// phases: load X -> qkv -> QK^T+bias+softmax (register) -> PV -> proj ->
//         channel-sum pooled into s_sum. 64KB LDS, regions phase-aliased.
// ---------------------------------------------------------------------------
__global__ __launch_bounds__(256) void k_attn(
    const __bf16* __restrict__ lnt,
    const __bf16* __restrict__ qwb, const float* __restrict__ qbias,
    const __bf16* __restrict__ pwb, const float* __restrict__ pbias,
    const float* __restrict__ rpe, float* __restrict__ s_sum) {
  __shared__ __align__(32) char smem[65536];
  __bf16* X  = (__bf16*)smem;            // 64x128 tokens; later probs[2][64][64]
  __bf16* Q  = (__bf16*)(smem + 16384);  // 64x128 q; later o_rm 64x128
  __bf16* Kt = (__bf16*)(smem + 32768);  // 64x128 k (rows are B^T rows for QK^T)
  __bf16* Vt = (__bf16*)(smem + 49152);  // 128x64 v^T (rows are B^T rows for PV)
  float*  tok = (float*)smem;            // 64 token sums (aliases X, phase 5)

  int blk = blockIdx.x;                  // 3136 windows
  int b  = blk / 196;
  int wi = blk - b * 196;
  int wh = wi / 14, ww = wi - (wi / 14) * 14;
  int tid = threadIdx.x;
  int lane = tid & 31, wid = tid >> 5;
  int ln15 = lane & 15;
  int rbase = (lane >> 4) << 3;
  v8f z8 = {0.f, 0.f, 0.f, 0.f, 0.f, 0.f, 0.f, 0.f};

  // ---- phase 1: load window tokens (zero rows for pad / out-of-image) ----
  {
    int row = tid & 63;
    int cc  = (tid >> 6) << 5;           // 0,32,64,96
    int rr  = row / 7, rc = row - (row / 7) * 7;
    int hh  = wh * 7 + rr, wp = ww * 7 + rc;
    uint4* dst = (uint4*)(X + row * 128 + cc);
    if (row < 49 && hh < 96 && wp < 96) {
      const uint4* src =
          (const uint4*)(lnt + ((size_t)(b * 9216 + hh * 96 + wp)) * 128 + cc);
      dst[0] = src[0]; dst[1] = src[1]; dst[2] = src[2]; dst[3] = src[3];
    } else {
      uint4 zz = {0u, 0u, 0u, 0u};
      dst[0] = zz; dst[1] = zz; dst[2] = zz; dst[3] = zz;
    }
  }
  __syncthreads();

  // ---- phase 2: qkv = X @ qkv_w^T + b ; wave handles 3 of 24 N-tiles ----
  {
    v8f qa[3][4];
#pragma unroll
    for (int nl = 0; nl < 3; ++nl)
#pragma unroll
      for (int mt = 0; mt < 4; ++mt) qa[nl][mt] = z8;
    for (int kc = 0; kc < 128; kc += 32) {
      v16bf a[4];
#pragma unroll
      for (int mt = 0; mt < 4; ++mt) a[mt] = load_a(X, 128, mt * 16, kc, lane);
#pragma unroll
      for (int nl = 0; nl < 3; ++nl) {
        v16bf bt = load_bt(qwb, 128, (wid * 3 + nl) * 16, kc, lane);
#pragma unroll
        for (int mt = 0; mt < 4; ++mt) qa[nl][mt] = wmma_bf16(a[mt], bt, qa[nl][mt]);
      }
    }
#pragma unroll
    for (int nl = 0; nl < 3; ++nl) {
      int o = (wid * 3 + nl) * 16 + ln15;
      float bia = qbias[o];
#pragma unroll
      for (int mt = 0; mt < 4; ++mt)
#pragma unroll
        for (int i = 0; i < 8; ++i) {
          int mm = mt * 16 + rbase + i;
          float val = qa[nl][mt][i] + bia;
          if (o < 128)       Q [mm * 128 + o]         = (__bf16)val;
          else if (o < 256)  Kt[mm * 128 + (o - 128)] = (__bf16)val;
          else               Vt[(o - 256) * 64 + mm]  = (__bf16)val;
        }
    }
  }
  __syncthreads();

  int head = wid >> 2;                   // wave = (head, mtile)
  int mt   = wid & 3;

  // ---- phase 3: scores = q@k^T/8 + bias, masked softmax in registers ----
  {
    v8f sc[4] = {z8, z8, z8, z8};
    for (int kc = 0; kc < 64; kc += 32) {
      v16bf a = load_a(Q + head * 64, 128, mt * 16, kc, lane);
#pragma unroll
      for (int nt = 0; nt < 4; ++nt) {
        v16bf bt = load_bt(Kt + head * 64, 128, nt * 16, kc, lane);
        sc[nt] = wmma_bf16(a, bt, sc[nt]);
      }
    }
    __bf16* P = (__bf16*)smem;           // probs[head][row][col]
#pragma unroll
    for (int i = 0; i < 8; ++i) {
      int qrow = mt * 16 + rbase + i;
      float vmax = -3.0e38f;
#pragma unroll
      for (int nt = 0; nt < 4; ++nt) {
        int col = nt * 16 + ln15;
        float sv = sc[nt][i] * 0.125f;   // 1/sqrt(64)
        if (col >= 49) {
          sv = -1.0e30f;
        } else if (qrow < 49) {
          int q0 = qrow / 7, c0 = col / 7;
          int d0 = q0 - c0 + 6;
          int d1 = (qrow - q0 * 7) - (col - c0 * 7) + 6;
          sv += rpe[(d0 * 13 + d1) * 2 + head];
        }
        sc[nt][i] = sv;
        vmax = fmaxf(vmax, sv);
      }
      vmax = fmaxf(vmax, __shfl_xor(vmax, 1, 16));
      vmax = fmaxf(vmax, __shfl_xor(vmax, 2, 16));
      vmax = fmaxf(vmax, __shfl_xor(vmax, 4, 16));
      vmax = fmaxf(vmax, __shfl_xor(vmax, 8, 16));
      float ssum = 0.f;
#pragma unroll
      for (int nt = 0; nt < 4; ++nt) {
        float e = __expf(sc[nt][i] - vmax);
        sc[nt][i] = e;
        ssum += e;
      }
      ssum += __shfl_xor(ssum, 1, 16);
      ssum += __shfl_xor(ssum, 2, 16);
      ssum += __shfl_xor(ssum, 4, 16);
      ssum += __shfl_xor(ssum, 8, 16);
      float rinv = 1.f / ssum;
#pragma unroll
      for (int nt = 0; nt < 4; ++nt)
        P[head * 4096 + qrow * 64 + nt * 16 + ln15] =
            (__bf16)(sc[nt][i] * rinv);
    }
  }
  __syncthreads();

  // ---- phase 4: o = probs @ v ----
  {
    const __bf16* P = (const __bf16*)smem;
    v8f oa[4] = {z8, z8, z8, z8};
    for (int kc = 0; kc < 64; kc += 32) {
      v16bf a = load_a(P + head * 4096, 64, mt * 16, kc, lane);
#pragma unroll
      for (int nt = 0; nt < 4; ++nt) {
        v16bf bt = load_bt(Vt + head * 64 * 64, 64, nt * 16, kc, lane);
        oa[nt] = wmma_bf16(a, bt, oa[nt]);
      }
    }
    __bf16* Orm = Q;                     // overwrite q region with o row-major
#pragma unroll
    for (int nt = 0; nt < 4; ++nt) {
      int n = head * 64 + nt * 16 + ln15;
#pragma unroll
      for (int i = 0; i < 8; ++i)
        Orm[(mt * 16 + rbase + i) * 128 + n] = (__bf16)oa[nt][i];
    }
  }
  __syncthreads();
  if (tid < 64) tok[tid] = 0.f;
  __syncthreads();

  // ---- phase 5: proj + per-token channel sums into tok[] ----
  {
    const __bf16* Orm = Q;
    int pmt = wid & 3;
    int pn0 = (wid >> 2) * 4;
    v8f pa[4] = {z8, z8, z8, z8};
    for (int kc = 0; kc < 128; kc += 32) {
      v16bf a = load_a(Orm, 128, pmt * 16, kc, lane);
#pragma unroll
      for (int nl = 0; nl < 4; ++nl) {
        v16bf bt = load_bt(pwb, 128, (pn0 + nl) * 16, kc, lane);
        pa[nl] = wmma_bf16(a, bt, pa[nl]);
      }
    }
#pragma unroll
    for (int nl = 0; nl < 4; ++nl) {
      int o = (pn0 + nl) * 16 + ln15;
      float pb = pbias[o];
#pragma unroll
      for (int i = 0; i < 8; ++i) {
        float val = pa[nl][i] + pb;
        val += __shfl_xor(val, 1, 16);
        val += __shfl_xor(val, 2, 16);
        val += __shfl_xor(val, 4, 16);
        val += __shfl_xor(val, 8, 16);
        if (ln15 == 0) atomicAdd(&tok[pmt * 16 + rbase + i], val);
      }
    }
  }
  __syncthreads();
  // pool valid tokens into s_sum[b][l], l = n/36
  if (tid < 49) {
    int rr = tid / 7, rc = tid - (tid / 7) * 7;
    int hh = wh * 7 + rr, wp = ww * 7 + rc;
    if (hh < 96 && wp < 96) {
      int n = hh * 96 + wp;
      atomicAdd(&s_sum[b * 256 + n / 36], tok[tid]);
    }
  }
}

// ---------------------------------------------------------------------------
// per-batch kurtosis / min / max of cs
// ---------------------------------------------------------------------------
__global__ __launch_bounds__(256) void k_stats(const float* __restrict__ cs,
                                               float* __restrict__ stats) {
  int b = blockIdx.x, t = threadIdx.x;
  __shared__ float r0[256], r1[256], r2[256];
  const float* p = cs + b * 9216;
  float s = 0.f, mn = 3e38f, mx = -3e38f;
  for (int i = t; i < 9216; i += 256) {
    float v = p[i];
    s += v; mn = fminf(mn, v); mx = fmaxf(mx, v);
  }
  r0[t] = s; r1[t] = mn; r2[t] = mx;
  __syncthreads();
  for (int o = 128; o; o >>= 1) {
    if (t < o) {
      r0[t] += r0[t + o];
      r1[t] = fminf(r1[t], r1[t + o]);
      r2[t] = fmaxf(r2[t], r2[t + o]);
    }
    __syncthreads();
  }
  float mu = r0[0] * (1.f / 9216.f);
  float cmin = r1[0], cmax = r2[0];
  __syncthreads();
  float s2 = 0.f, s4 = 0.f;
  for (int i = t; i < 9216; i += 256) {
    float d = p[i] - mu;
    float d2 = d * d;
    s2 += d2; s4 += d2 * d2;
  }
  r0[t] = s2; r1[t] = s4;
  __syncthreads();
  for (int o = 128; o; o >>= 1) {
    if (t < o) { r0[t] += r0[t + o]; r1[t] += r1[t + o]; }
    __syncthreads();
  }
  if (t == 0) {
    float n = 9216.f;
    float m2 = r0[0], m4 = r1[0];
    float var = m2 / (n - 1.f);
    float kurt = n * (n + 1.f) / ((n - 1.f) * (n - 2.f) * (n - 3.f)) *
                     m4 / (var * var) -
                 3.f * (n - 1.f) * (n - 1.f) / ((n - 2.f) * (n - 3.f));
    stats[b * 4 + 0] = fabsf(kurt);
    stats[b * 4 + 1] = cmin;
    stats[b * 4 + 2] = cmax;
    stats[b * 4 + 3] = (cmax - cmin) * (1.f / 256.f);  // inter
  }
}

// ---------------------------------------------------------------------------
// quant tensor (B,9216,256) -> d_out tail, and sta = sum over N
// ---------------------------------------------------------------------------
__global__ __launch_bounds__(256) void k_quant(const float* __restrict__ cs,
                                               const float* __restrict__ stats,
                                               float* __restrict__ outq,
                                               float* __restrict__ sta_sum) {
  int b  = blockIdx.y;
  int n0 = blockIdx.x * 16;
  int l  = threadIdx.x;
  float Kabs = stats[b * 4 + 0];
  float cmin = stats[b * 4 + 1];
  float cmax = stats[b * 4 + 2];
  float inter = stats[b * 4 + 3];
  float ql = (2.f * l + 1.f) * (1.f / 512.f) * (cmax - cmin) + cmin;
  float thr = 1.f - inter;
  float acc = 0.f;
  for (int j = 0; j < 16; ++j) {
    int n = n0 + j;
    float c = cs[b * 9216 + n];
    float q = 1.f - fabsf(ql - c);
    float val = Kabs * __expf(-q);
    val = (val > thr) ? val : 0.f;
    outq[((size_t)(b * 9216 + n) << 8) + l] = val;
    acc += val;
  }
  atomicAdd(&sta_sum[b * 256 + l], acc);
}

// ---------------------------------------------------------------------------
// head MLP: sta(2) -> f1(64) leaky -> f2(128) relu -> out(256) + BN + relu
// ---------------------------------------------------------------------------
__global__ __launch_bounds__(128) void k_head(
    const float* __restrict__ s_sum, const float* __restrict__ sta_sum,
    const float* __restrict__ f1w, const float* __restrict__ f2w,
    const float* __restrict__ outw,
    const float* __restrict__ og, const float* __restrict__ ob,
    const float* __restrict__ om, const float* __restrict__ ov,
    const float* __restrict__ gate, float* __restrict__ outh) {
  int b = blockIdx.y, l = blockIdx.x, t = threadIdx.x;
  __shared__ float h1[64], h2[128];
  float c0 = gate[0] * s_sum[b * 256 + l] * (1.f / (128.f * 36.f));
  float c1 = sta_sum[b * 256 + l];
  if (t < 64) {
    float v = f1w[t * 2 + 0] * c0 + f1w[t * 2 + 1] * c1;
    h1[t] = v > 0.f ? v : 0.01f * v;
  }
  __syncthreads();
  {
    float a = 0.f;
    for (int j = 0; j < 64; ++j) a += f2w[t * 64 + j] * h1[j];
    h2[t] = fmaxf(a, 0.f);
  }
  __syncthreads();
  for (int o = t; o < 256; o += 128) {
    float a = 0.f;
    for (int j = 0; j < 128; ++j) a += outw[o * 128 + j] * h2[j];
    a = (a - om[o]) * rsqrtf(ov[o] + 1e-5f) * og[o] + ob[o];
    outh[((size_t)b * 256 + o) * 256 + l] = fmaxf(a, 0.f);
  }
}

// ---------------------------------------------------------------------------
extern "C" void kernel_launch(void* const* d_in, const int* in_sizes, int n_in,
                              void* d_out, int out_size, void* d_ws,
                              size_t ws_size, hipStream_t stream) {
  (void)in_sizes; (void)n_in; (void)out_size; (void)ws_size;
  const float* x       = (const float*)d_in[0];
  const float* conv1_w = (const float*)d_in[1];
  const float* bn1_g   = (const float*)d_in[2];
  const float* bn1_b   = (const float*)d_in[3];
  const float* bn1_m   = (const float*)d_in[4];
  const float* bn1_v   = (const float*)d_in[5];
  const float* ln_g    = (const float*)d_in[6];
  const float* ln_b    = (const float*)d_in[7];
  const float* qkv_w   = (const float*)d_in[8];
  const float* qkv_b   = (const float*)d_in[9];
  const float* proj_w  = (const float*)d_in[10];
  const float* proj_b  = (const float*)d_in[11];
  const float* rpe     = (const float*)d_in[12];
  const float* conv2_w = (const float*)d_in[13];
  const float* f1_w    = (const float*)d_in[14];
  const float* f2_w    = (const float*)d_in[15];
  const float* out_w   = (const float*)d_in[16];
  const float* obn_g   = (const float*)d_in[17];
  const float* obn_b   = (const float*)d_in[18];
  const float* obn_m   = (const float*)d_in[19];
  const float* obn_v   = (const float*)d_in[20];
  const float* gate    = (const float*)d_in[21];

  float* outh = (float*)d_out;                  // (16,256,256)
  float* outq = outh + (size_t)16 * 256 * 256;  // (16,9216,256)

  // workspace carve-out (256B aligned)
  char* w = (char*)d_ws;
  size_t off = 0;
  auto carve = [&](size_t bytes) -> char* {
    char* p = w + off;
    off = (off + bytes + 255) & ~(size_t)255;
    return p;
  };
  __bf16* lnt     = (__bf16*)carve((size_t)16 * 9216 * 128 * 2);
  float*  cs      = (float*)carve((size_t)147456 * 4);
  float*  s_sum   = (float*)carve(4096 * 4);
  float*  sta_sum = (float*)carve(4096 * 4);
  float*  stats   = (float*)carve(64 * 4);
  __bf16* c1b     = (__bf16*)carve((size_t)294912 * 2);
  __bf16* qb      = (__bf16*)carve((size_t)49152 * 2);
  __bf16* pb      = (__bf16*)carve((size_t)16384 * 2);

  k_prep<<<1152, 256, 0, stream>>>(conv1_w, qkv_w, proj_w, c1b, qb, pb,
                                   s_sum, sta_sum);
  k_conv1<<<4608, 128, 0, stream>>>(x, c1b, bn1_g, bn1_b, bn1_m, bn1_v,
                                    ln_g, ln_b, conv2_w, lnt, cs);
  k_attn<<<3136, 256, 0, stream>>>(lnt, qb, qkv_b, pb, proj_b, rpe, s_sum);
  k_stats<<<16, 256, 0, stream>>>(cs, stats);
  k_quant<<<dim3(576, 16), 256, 0, stream>>>(cs, stats, outq, sta_sum);
  k_head<<<dim3(256, 16), 128, 0, stream>>>(s_sum, sta_sum, f1_w, f2_w, out_w,
                                            obn_g, obn_b, obn_m, obn_v, gate,
                                            outh);
}
